// MatmulAttention_25494925869080
// MI455X (gfx1250) — matrix operand order repaired and checked by reading, not compiled, not run on hardware
//
#include <hip/hip_runtime.h>
#include <hip/hip_bf16.h>

// ---- problem constants (fixed by the reference) ----
#define BATCH   2
#define HEADS   16
#define SEQ     2048
#define HD      64              // head_dim
#define KT      64              // keys per LDS tile
#define LS      72              // LDS row stride in halves (16B-aligned rows, conflict padding)
#define WAVES   4               // 128 threads / wave32
#define QW      32              // Q rows per wave (two 16-row A tiles)

typedef __attribute__((ext_vector_type(16))) _Float16 v16h;
typedef __attribute__((ext_vector_type(8)))  _Float16 v8h;
typedef __attribute__((ext_vector_type(4)))  _Float16 v4h;
typedef __attribute__((ext_vector_type(8)))  float    v8f;

union Frag16 { v16h v; v8h h[2]; _Float16 e[16]; };
union Acc8   { v8f  v; float f[8]; };

// ---- 16-lane-group xor shuffle on the VALU (v_permlane16) instead of ds_bpermute ----
__device__ __forceinline__ float lane16_perm(float v, int sel0, int sel1) {
#if __has_builtin(__builtin_amdgcn_permlane16)
    int i = __builtin_bit_cast(int, v);
    i = __builtin_amdgcn_permlane16(i, i, sel0, sel1, false, false);
    return __builtin_bit_cast(float, i);
#else
    (void)sel0; (void)sel1;
    return v;
#endif
}

__device__ __forceinline__ float rowmax16(float v) {
#if __has_builtin(__builtin_amdgcn_permlane16)
    v = fmaxf(v, lane16_perm(v, 0x67452301, 0xEFCDAB89));   // xor 1
    v = fmaxf(v, lane16_perm(v, 0x54761032, 0xDCFE98BA));   // xor 2
    v = fmaxf(v, lane16_perm(v, 0x32107654, 0xBA98FEDC));   // xor 4
    v = fmaxf(v, lane16_perm(v, 0xFEDCBA98, 0x76543210));   // xor 8
#else
    #pragma unroll
    for (int off = 8; off >= 1; off >>= 1) v = fmaxf(v, __shfl_xor(v, off, 32));
#endif
    return v;
}

__device__ __forceinline__ float rowsum16(float v) {
#if __has_builtin(__builtin_amdgcn_permlane16)
    v += lane16_perm(v, 0x67452301, 0xEFCDAB89);            // xor 1
    v += lane16_perm(v, 0x54761032, 0xDCFE98BA);            // xor 2
    v += lane16_perm(v, 0x32107654, 0xBA98FEDC);            // xor 4
    v += lane16_perm(v, 0xFEDCBA98, 0x76543210);            // xor 8
#else
    #pragma unroll
    for (int off = 8; off >= 1; off >>= 1) v += __shfl_xor(v, off, 32);
#endif
    return v;
}

__global__ __launch_bounds__(128) void
attn_flash_wmma_f16(const float* __restrict__ hs, float* __restrict__ out)
{
    // Q=K=V: one f16 tile serves as K and V. Row-major copy (K^T B-fragments)
    // plus transposed copy (V B-fragments) -> all WMMA fragment reads are
    // contiguous ds_load_b128. B-fragments are reused by TWO Q tiles per wave.
    __shared__ _Float16 ldsKV[KT * LS];            //  9216 B  [key][d]
    __shared__ _Float16 ldsVT[HD * LS];            //  9216 B  [d][key]
    __shared__ _Float16 ldsP [WAVES * QW * LS];    // 18432 B  per-wave P tile [qrow(32)][key]

    const int tid  = threadIdx.x;
    const int wave = tid >> 5;
    const int lane = tid & 31;
    const int lh   = lane >> 4;      // half-wave id: 0 -> lanes 0-15, 1 -> lanes 16-31
    const int ln   = lane & 15;

    const int qtiles = SEQ / (WAVES * QW);       // 16 q-blocks of 128 rows
    const int bh     = blockIdx.x / qtiles;      // fused batch*head
    const int qblk   = blockIdx.x % qtiles;
    const int qbase  = qblk * (WAVES * QW) + wave * QW;   // wave owns rows [qbase, qbase+32)

    const float* __restrict__ src = hs + (size_t)bh * SEQ * HD;

    // ---- Q fragments: two 16x64 A tiles, each as two 16x32 f16 chunks ----
    // A layout: lanes 0-15 hold K {c*32+0..7, c*32+16..23}; lanes 16-31 {+8, +8}
    Frag16 qa[2][2];                 // [q-tile t][k-chunk c]
    #pragma unroll
    for (int t = 0; t < 2; ++t) {
        const float* qptr = src + (size_t)(qbase + t * 16 + ln) * HD;
        #pragma unroll
        for (int c = 0; c < 2; ++c) {
            const int base = 32 * c + 8 * lh;
            const float4* p0 = (const float4*)(qptr + base);        // 32B aligned
            const float4* p1 = (const float4*)(qptr + base + 16);
            float4 a0 = p0[0], a1 = p0[1], b0 = p1[0], b1 = p1[1];
            qa[t][c].e[0]  = (_Float16)a0.x; qa[t][c].e[1]  = (_Float16)a0.y;
            qa[t][c].e[2]  = (_Float16)a0.z; qa[t][c].e[3]  = (_Float16)a0.w;
            qa[t][c].e[4]  = (_Float16)a1.x; qa[t][c].e[5]  = (_Float16)a1.y;
            qa[t][c].e[6]  = (_Float16)a1.z; qa[t][c].e[7]  = (_Float16)a1.w;
            qa[t][c].e[8]  = (_Float16)b0.x; qa[t][c].e[9]  = (_Float16)b0.y;
            qa[t][c].e[10] = (_Float16)b0.z; qa[t][c].e[11] = (_Float16)b0.w;
            qa[t][c].e[12] = (_Float16)b1.x; qa[t][c].e[13] = (_Float16)b1.y;
            qa[t][c].e[14] = (_Float16)b1.z; qa[t][c].e[15] = (_Float16)b1.w;
        }
    }

    // ---- flash-attention running state (f32) ----
    Acc8 acc[2][4];                  // [q-tile][d-column tile]
    #pragma unroll
    for (int t = 0; t < 2; ++t)
        #pragma unroll
        for (int nd = 0; nd < 4; ++nd)
            #pragma unroll
            for (int i = 0; i < 8; ++i) acc[t][nd].f[i] = 0.0f;

    float mrow[2][8], lrow[2][8];    // per-lane rows r -> query row (t*16 + r + 8*lh)
    #pragma unroll
    for (int t = 0; t < 2; ++t)
        #pragma unroll
        for (int r = 0; r < 8; ++r) { mrow[t][r] = -3.0e38f; lrow[t][r] = 0.0f; }

    _Float16* myP = ldsP + wave * QW * LS;

    for (int kt = 0; kt < SEQ; kt += KT) {
        // ---- cooperative tile load: 64x64 f32 -> f16 LDS, row-major + transposed ----
        {
            const float4* src4 = (const float4*)(src + (size_t)kt * HD);
            #pragma unroll
            for (int j = 0; j < 8; ++j) {
                const int flat4 = tid + j * 128;        // 1024 float4 per tile
                const int row   = flat4 >> 4;           // 16 float4 per 64-elem row
                const int col   = (flat4 & 15) * 4;
                float4 f = src4[flat4];
                v4h h; h[0] = (_Float16)f.x; h[1] = (_Float16)f.y;
                       h[2] = (_Float16)f.z; h[3] = (_Float16)f.w;
                *(v4h*)(ldsKV + row * LS + col) = h;    // one ds_store_b64
                ldsVT[(col + 0) * LS + row] = h[0];     // transposed copy (scattered b16)
                ldsVT[(col + 1) * LS + row] = h[1];
                ldsVT[(col + 2) * LS + row] = h[2];
                ldsVT[(col + 3) * LS + row] = h[3];
            }
        }
        __syncthreads();

        // prefetch next tile into cache while we compute (global_prefetch_b8)
        if (kt + KT < SEQ)
            __builtin_prefetch(src + (size_t)(kt + KT) * HD + tid * 32, 0, 0);

        // ---- scores: S = Q * K^T; each K^T B-fragment feeds BOTH Q tiles ----
        Acc8 sc[2][4];
        #pragma unroll
        for (int t = 0; t < 2; ++t)
            #pragma unroll
            for (int n = 0; n < 4; ++n)
                #pragma unroll
                for (int i = 0; i < 8; ++i) sc[t][n].f[i] = 0.0f;

        #pragma unroll
        for (int n = 0; n < 4; ++n) {
            #pragma unroll
            for (int c = 0; c < 2; ++c) {
                // B-frag of K^T: lane holds key column (n*16+ln); k = 32c + 8*lh + {0..7, 16..23}
                Frag16 b;
                const _Float16* kp = ldsKV + (n * 16 + ln) * LS + 32 * c + 8 * lh;
                b.h[0] = *(const v8h*)(kp);
                b.h[1] = *(const v8h*)(kp + 16);
                sc[0][n].v = __builtin_amdgcn_wmma_f32_16x16x32_f16(
                    false, qa[0][c].v, false, b.v, (short)0, sc[0][n].v, false, false);
                sc[1][n].v = __builtin_amdgcn_wmma_f32_16x16x32_f16(
                    false, qa[1][c].v, false, b.v, (short)0, sc[1][n].v, false, false);
            }
        }

        // ---- online softmax over this 64-key slab (both q-tiles) ----
        // C-layout: lane (lh,ln) VGPR r  <->  row (r + 8*lh), col ln
        #pragma unroll
        for (int t = 0; t < 2; ++t) {
            #pragma unroll
            for (int r = 0; r < 8; ++r) {
                float tmax = sc[t][0].f[r];
                #pragma unroll
                for (int n = 1; n < 4; ++n) tmax = fmaxf(tmax, sc[t][n].f[r]);
                tmax = rowmax16(tmax);

                const float newm  = fmaxf(mrow[t][r], tmax);
                const float scale = __expf(mrow[t][r] - newm);
                mrow[t][r] = newm;

                float p[4], rsum = 0.0f;
                #pragma unroll
                for (int n = 0; n < 4; ++n) { p[n] = __expf(sc[t][n].f[r] - newm); rsum += p[n]; }
                rsum = rowsum16(rsum);
                lrow[t][r] = lrow[t][r] * scale + rsum;

                #pragma unroll
                for (int nd = 0; nd < 4; ++nd) acc[t][nd].f[r] *= scale;

                // stage P (f16) to LDS to re-layout C-format -> A-format
                const int prow = t * 16 + r + 8 * lh;
                #pragma unroll
                for (int n = 0; n < 4; ++n)
                    myP[prow * LS + n * 16 + ln] = (_Float16)p[n];
            }
        }
        __syncthreads();

        // ---- O += P * V; each V B-fragment feeds BOTH Q tiles ----
        #pragma unroll
        for (int c = 0; c < 2; ++c) {
            // A-frags of P: row = ln within each tile; keys {32c+8lh+0..7, +16..23}
            Frag16 a0, a1;
            {
                const _Float16* pp0 = myP + ln * LS + 32 * c + 8 * lh;
                const _Float16* pp1 = myP + (16 + ln) * LS + 32 * c + 8 * lh;
                a0.h[0] = *(const v8h*)(pp0);
                a0.h[1] = *(const v8h*)(pp0 + 16);
                a1.h[0] = *(const v8h*)(pp1);
                a1.h[1] = *(const v8h*)(pp1 + 16);
            }
            #pragma unroll
            for (int nd = 0; nd < 4; ++nd) {
                // B-frag of V from transposed copy: lane holds d-column nd*16+ln;
                // k = key = 32c + 8lh + {0..7, 16..23} -> two 8-half runs in ldsVT row
                Frag16 bv;
                const _Float16* vp = ldsVT + (nd * 16 + ln) * LS + 32 * c + 8 * lh;
                bv.h[0] = *(const v8h*)(vp);
                bv.h[1] = *(const v8h*)(vp + 16);
                acc[0][nd].v = __builtin_amdgcn_wmma_f32_16x16x32_f16(
                    false, a0.v, false, bv.v, (short)0, acc[0][nd].v, false, false);
                acc[1][nd].v = __builtin_amdgcn_wmma_f32_16x16x32_f16(
                    false, a1.v, false, bv.v, (short)0, acc[1][nd].v, false, false);
            }
        }
        __syncthreads();   // protect ldsKV/ldsVT before next tile load
    }

    // ---- epilogue: O / l, f32 store ----
    float* obase = out + (size_t)bh * SEQ * HD;
    #pragma unroll
    for (int t = 0; t < 2; ++t) {
        #pragma unroll
        for (int r = 0; r < 8; ++r) {
            const int   row = qbase + t * 16 + r + 8 * lh;
            const float inv = 1.0f / lrow[t][r];
            #pragma unroll
            for (int nd = 0; nd < 4; ++nd)
                obase[(size_t)row * HD + nd * 16 + ln] = acc[t][nd].f[r] * inv;
        }
    }
}

extern "C" void kernel_launch(void* const* d_in, const int* in_sizes, int n_in,
                              void* d_out, int out_size, void* d_ws, size_t ws_size,
                              hipStream_t stream)
{
    (void)in_sizes; (void)n_in; (void)out_size; (void)d_ws; (void)ws_size;
    const float* hs  = (const float*)d_in[0];
    float*       out = (float*)d_out;

    const int qtiles = SEQ / (WAVES * QW);              // 16
    dim3 grid(BATCH * HEADS * qtiles);                  // 512 workgroups
    dim3 block(WAVES * 32);                             // 128 threads = 4 wave32
    attn_flash_wmma_f16<<<grid, block, 0, stream>>>(hs, out);
}
